// NNAttention_25374666785049
// MI455X (gfx1250) — matrix operand order repaired and checked by reading, not compiled, not run on hardware
//
#include <hip/hip_runtime.h>
#include <hip/hip_bf16.h>

typedef __attribute__((ext_vector_type(16))) __bf16 v16bf;
typedef __attribute__((ext_vector_type(8)))  float  v8f;

#define B_   8
#define S_   2048
#define D_   512   // DIN == DOUT

__device__ __forceinline__ v8f wmma_bf16(v16bf a, v16bf b, v8f c) {
  return __builtin_amdgcn_wmma_f32_16x16x32_bf16(false, a, false, b,
                                                 (short)0, c, false, false);
}

// 16-bit A-operand layout (ISA 7.12.2, 16x32 MxK): per lane (m = lane&15,
// hi = lane>>4) the 16 elements are two contiguous 8-element runs:
//   e in [0,8):  k = hi*8 + e          e in [8,16): k = 16 + hi*8 + (e-8)
// -> exactly two 16-byte loads per operand.
union v16bf_u { v16bf v; uint4 q[2]; };

// ---------------------------------------------------------------------------
// Kernel 1: convert + transpose weights to bf16: Wt[w][n][k] = (bf16)W[w][k][n]
// ---------------------------------------------------------------------------
__global__ void wt_cvt_kernel(const float* __restrict__ Wq,
                              const float* __restrict__ Wk,
                              const float* __restrict__ Wv,
                              __bf16* __restrict__ Wt) {
  int idx = blockIdx.x * blockDim.x + threadIdx.x;
  if (idx >= 3 * D_ * D_) return;
  int w = idx / (D_ * D_);
  int r = idx % (D_ * D_);
  int n = r / D_;
  int k = r % D_;
  const float* W = (w == 0) ? Wq : (w == 1) ? Wk : Wv;
  Wt[(size_t)w * D_ * D_ + (size_t)n * D_ + k] = (__bf16)W[(size_t)k * D_ + n];
}

// ---------------------------------------------------------------------------
// Kernel 2: QKV projection. Activation tile converted to bf16 in LDS once
// (shared by all 8 waves); each wave owns a 16x16 output tile.
// q,k stored row-major [B*S][D]; v stored TRANSPOSED vt[b][d][s] so the
// attention P.V B-operand becomes contiguous global_load_b128s.
// grid = (B*S/16, DOUT/128, 3), block = 256.
// ---------------------------------------------------------------------------
__global__ __launch_bounds__(256)
void qkv_proj_kernel(const float* __restrict__ Q, const float* __restrict__ K,
                     const float* __restrict__ V,
                     const __bf16* __restrict__ Wt,
                     const float* __restrict__ bq, const float* __restrict__ bk,
                     const float* __restrict__ bv,
                     __bf16* __restrict__ qb, __bf16* __restrict__ kb,
                     __bf16* __restrict__ vt) {
  __shared__ __align__(16) __bf16 xa[16 * D_];   // 16 KB bf16 activation tile

  const int z = blockIdx.z;
  const float*  X    = (z == 0) ? Q  : (z == 1) ? K  : V;
  const float*  bias = (z == 0) ? bq : (z == 1) ? bk : bv;
  const __bf16* W    = Wt + (size_t)z * D_ * D_;

  const int lane = threadIdx.x & 31;
  const int wave = threadIdx.x >> 5;
  const int lo = lane & 15;
  const int hi = lane >> 4;
  const int row0 = blockIdx.x * 16;
  const int n = blockIdx.y * 128 + wave * 16 + lo;

  // cooperative f32 -> bf16 of the 16x512 activation tile
  {
    const float* Xr = X + (size_t)row0 * D_;
    for (int i = threadIdx.x * 4; i < 16 * D_; i += blockDim.x * 4) {
      float4 f = *(const float4*)(Xr + i);
      union { __bf16 h[4]; uint2 u; } t;
      t.h[0] = (__bf16)f.x; t.h[1] = (__bf16)f.y;
      t.h[2] = (__bf16)f.z; t.h[3] = (__bf16)f.w;
      *(uint2*)(xa + i) = t.u;
    }
  }
  __syncthreads();

  v8f c = {};
  const __bf16* wrow = W + (size_t)n * D_;
  for (int d0 = 0; d0 < D_; d0 += 32) {
    v16bf_u a, b;
    a.q[0] = *(const uint4*)(xa + lo * D_ + d0 + hi * 8);
    a.q[1] = *(const uint4*)(xa + lo * D_ + d0 + 16 + hi * 8);
    b.q[0] = *(const uint4*)(wrow + d0 + hi * 8);
    b.q[1] = *(const uint4*)(wrow + d0 + 16 + hi * 8);
    c = wmma_bf16(a.v, b.v, c);
  }
  const float bn = bias[n];

  if (z == 2) {
    // V: store transposed vt[b][n][s]; rows r map to 8 consecutive s values
    const int bb = row0 / S_;
    const int s  = row0 % S_ + 8 * hi;
    union { __bf16 h[8]; uint4 u; } tv;
#pragma unroll
    for (int r = 0; r < 8; ++r) tv.h[r] = (__bf16)(c[r] + bn);
    *(uint4*)(vt + ((size_t)bb * D_ + n) * S_ + s) = tv.u;
  } else {
    __bf16* dst = (z == 0) ? qb : kb;
#pragma unroll
    for (int r = 0; r < 8; ++r) {
      int mr = r + 8 * hi;                    // C/D layout: row = vgpr + 8*hi
      dst[(size_t)(row0 + mr) * D_ + n] = (__bf16)(c[r] + bn);
    }
  }
}

// ---------------------------------------------------------------------------
// Kernel 3: flash attention, one block per (batch, 16 q-rows), 8 waves.
// Per 128-key chunk: wave w computes the 16x16 score tile for keys
// [kc+16w, kc+16w+16) (K read straight from global/L2, 2xb128 per lane);
// scores shared via LDS; softmax row-partitioned (2 rows/wave); every wave
// then does P(16x128) . V(128x64) for its own 64 output columns with V
// fetched from the transposed vt (2xb128 per lane). Zero redundant WMMAs.
// ---------------------------------------------------------------------------
__global__ __launch_bounds__(256)
void flash_attn_kernel(const __bf16* __restrict__ qb,
                       const __bf16* __restrict__ kb,
                       const __bf16* __restrict__ vt,
                       float* __restrict__ out) {
  __shared__ __align__(16) __bf16 qs[16 * D_];    // 16 KB q tile
  __shared__ __align__(16) float  sc[16 * 128];   //  8 KB raw scores
  __shared__ __align__(16) __bf16 pp[16 * 128];   //  4 KB probabilities (A layout)
  __shared__ float lm[16], ll[16], sscale[16];

  const int lane = threadIdx.x & 31;
  const int wave = threadIdx.x >> 5;
  const int lo = lane & 15;
  const int hi = lane >> 4;

  const int b  = blockIdx.x / (S_ / 16);
  const int s0 = (blockIdx.x % (S_ / 16)) * 16;
  const size_t qrow = (size_t)b * S_ + s0;
  const int wd0 = wave * 64;

  // stage q tile (contiguous 16 KB)
  {
    const uint4* src = (const uint4*)(qb + qrow * D_);
    uint4* dq = (uint4*)qs;
    for (int i = threadIdx.x; i < 16 * D_ / 8; i += blockDim.x) dq[i] = src[i];
  }
  if (threadIdx.x < 16) { lm[threadIdx.x] = -3.0e38f; ll[threadIdx.x] = 0.0f; }
  __syncthreads();

  v8f acc[4] = {};
  const __bf16* kbase = kb + (size_t)b * S_ * D_;
  const __bf16* vbase = vt + (size_t)b * D_ * S_;

  for (int kc = 0; kc < S_; kc += 128) {
    // ---- scores: this wave's 16 keys, f32 accumulate over d=512 ----
    v8f c = {};
    const __bf16* krow = kbase + (size_t)(kc + wave * 16 + lo) * D_;
    for (int d0 = 0; d0 < D_; d0 += 32) {
      v16bf_u a, bk_;
      a.q[0]   = *(const uint4*)(qs + lo * D_ + d0 + hi * 8);
      a.q[1]   = *(const uint4*)(qs + lo * D_ + d0 + 16 + hi * 8);
      bk_.q[0] = *(const uint4*)(krow + d0 + hi * 8);
      bk_.q[1] = *(const uint4*)(krow + d0 + 16 + hi * 8);
      c = wmma_bf16(a.v, bk_.v, c);
    }
#pragma unroll
    for (int r = 0; r < 8; ++r)
      sc[(r + 8 * hi) * 128 + wave * 16 + lo] = c[r];
    __syncthreads();

    // ---- online softmax: wave handles rows 2*wave, 2*wave+1 (4 cols/lane) ----
#pragma unroll
    for (int rr = 0; rr < 2; ++rr) {
      const int m = wave * 2 + rr;
      const float mold = lm[m];
      float x0 = sc[m * 128 + lane * 4 + 0];
      float x1 = sc[m * 128 + lane * 4 + 1];
      float x2 = sc[m * 128 + lane * 4 + 2];
      float x3 = sc[m * 128 + lane * 4 + 3];
      float vm = fmaxf(fmaxf(x0, x1), fmaxf(x2, x3));
#pragma unroll
      for (int off = 16; off > 0; off >>= 1)
        vm = fmaxf(vm, __shfl_xor(vm, off, 32));
      const float mn = fmaxf(mold, vm);
      float p0 = __expf(x0 - mn), p1 = __expf(x1 - mn);
      float p2 = __expf(x2 - mn), p3 = __expf(x3 - mn);
      union { __bf16 h[4]; uint2 u; } up;
      up.h[0] = (__bf16)p0; up.h[1] = (__bf16)p1;
      up.h[2] = (__bf16)p2; up.h[3] = (__bf16)p3;
      *(uint2*)(pp + m * 128 + lane * 4) = up.u;
      float sum = (p0 + p1) + (p2 + p3);
#pragma unroll
      for (int off = 16; off > 0; off >>= 1)
        sum += __shfl_xor(sum, off, 32);
      if (lane == 0) {
        const float scl = __expf(mold - mn);
        sscale[m] = scl;
        ll[m] = ll[m] * scl + sum;
        lm[m] = mn;
      }
    }
    __syncthreads();

    // ---- rescale accumulators, then acc += P(16x128) . V(128x64) ----
#pragma unroll
    for (int r = 0; r < 8; ++r) {
      const float scl = sscale[r + 8 * hi];
#pragma unroll
      for (int t = 0; t < 4; ++t) acc[t][r] *= scl;
    }
#pragma unroll
    for (int kk = 0; kk < 128; kk += 32) {
      v16bf_u ap;
      ap.q[0] = *(const uint4*)(pp + lo * 128 + kk + hi * 8);
      ap.q[1] = *(const uint4*)(pp + lo * 128 + kk + 16 + hi * 8);
#pragma unroll
      for (int t = 0; t < 4; ++t) {
        const __bf16* vcol = vbase + (size_t)(wd0 + t * 16 + lo) * S_ + kc + kk;
        v16bf_u bv_;
        bv_.q[0] = *(const uint4*)(vcol + hi * 8);
        bv_.q[1] = *(const uint4*)(vcol + 16 + hi * 8);
        acc[t] = wmma_bf16(ap.v, bv_.v, acc[t]);
      }
    }
    __syncthreads();   // protect sc/pp/lm/ll/sscale for next chunk
  }

  // ---- normalize and write f32 output ----
#pragma unroll
  for (int r = 0; r < 8; ++r) {
    const int mr = r + 8 * hi;
    const float inv = 1.0f / ll[mr];
#pragma unroll
    for (int t = 0; t < 4; ++t)
      out[(qrow + mr) * D_ + wd0 + t * 16 + lo] = acc[t][r] * inv;
  }
}

// ---------------------------------------------------------------------------
extern "C" void kernel_launch(void* const* d_in, const int* in_sizes, int n_in,
                              void* d_out, int out_size, void* d_ws, size_t ws_size,
                              hipStream_t stream) {
  const float* Q  = (const float*)d_in[0];
  const float* K  = (const float*)d_in[1];
  const float* V  = (const float*)d_in[2];
  const float* Wq = (const float*)d_in[3];
  const float* bq = (const float*)d_in[4];
  const float* Wk = (const float*)d_in[5];
  const float* bk = (const float*)d_in[6];
  const float* Wv = (const float*)d_in[7];
  const float* bv = (const float*)d_in[8];

  char* ws = (char*)d_ws;
  __bf16* Wt = (__bf16*)ws;                                   // 3 * 512*512 bf16
  __bf16* qb = (__bf16*)(ws + (size_t)3 * D_ * D_ * 2);       // [B*S][D] bf16
  __bf16* kb = qb + (size_t)B_ * S_ * D_;                     // [B*S][D] bf16
  __bf16* vt = kb + (size_t)B_ * S_ * D_;                     // [B][D][S] bf16

  wt_cvt_kernel<<<(3 * D_ * D_ + 255) / 256, 256, 0, stream>>>(Wq, Wk, Wv, Wt);

  dim3 gproj(B_ * S_ / 16, D_ / 128, 3);
  qkv_proj_kernel<<<gproj, 256, 0, stream>>>(Q, K, V, Wt, bq, bk, bv, qb, kb, vt);

  flash_attn_kernel<<<B_ * S_ / 16, 256, 0, stream>>>(qb, kb, vt, (float*)d_out);
}
